// IntBlock_3521873183453
// MI455X (gfx1250) — 8-bit matrix operand order repaired and checked by reading, compiled, not run on hardware
//
#include <hip/hip_runtime.h>


typedef int v8i __attribute__((ext_vector_type(8)));

#define CIN    128
#define LLEN   4096
#define COUT   256
#define TILE_L 128
#define STR    144

__device__ __forceinline__ int dw_shift_clamp(int v) {
  int a = v < 0 ? -v : v;
  v = (a >= 4) ? v : 0;
  v >>= 2;
  v = v < -128 ? -128 : v;
  v = v >  127 ?  127 : v;
  return v;
}

__global__ __launch_bounds__(256)
void fused_dwpw_wmma_kernel(const int* __restrict__ x,
                            const int* __restrict__ w_dw,
                            const int* __restrict__ w_pw,
                            float* __restrict__ out) {
  __shared__ __align__(16) signed char lds_pool[TILE_L * STR + COUT * STR];
  signed char* const hl  = lds_pool;
  signed char* const wl  = lds_pool + TILE_L * STR;
  signed char* const stg = wl;

  const int tid  = threadIdx.x;
  const int lane = tid & 31;
  const int wave = tid >> 5;
  const int tile = blockIdx.x & 31;
  const int b    = blockIdx.x >> 5;
  const int l0   = tile * TILE_L;

  for (int idx = tid; idx < COUT * (CIN / 4); idx += 256) {
    const int o  = idx >> 5;
    const int cw = (idx & 31) * 4;
    const int* wp = w_pw + o * CIN + cw;
    const int packed = (wp[0] & 0xFF)        | ((wp[1] & 0xFF) << 8) |
                       ((wp[2] & 0xFF) << 16) | ((wp[3] & 0xFF) << 24);
    *(int*)(&wl[o * STR + cw]) = packed;
  }

  {
    const int c   = tid >> 1;
    const int seg = tid & 1;
    const int ls  = l0 + seg * 64;
    const int* xr = x + ((size_t)b * CIN + c) * LLEN;

    __builtin_prefetch(xr + ls, 0, 1);

    int w[7];
    #pragma unroll
    for (int k = 0; k < 7; ++k) w[k] = w_dw[c * 7 + k];

    const bool interior = (tile >= 1) && (tile <= 30);

    for (int ii = 0; ii < 64; ii += 4) {
      const int base = ls + ii - 4;
      int v[12];
      if (interior) {
        #pragma unroll
        for (int j = 0; j < 3; ++j) {
          const int4 q = *(const int4*)(xr + base + j * 4);
          v[j*4+0] = q.x; v[j*4+1] = q.y; v[j*4+2] = q.z; v[j*4+3] = q.w;
        }
      } else {
        #pragma unroll
        for (int j = 0; j < 3; ++j) {
          const int p = base + j * 4;
          if (p >= 0 && p < LLEN) {
            const int4 q = *(const int4*)(xr + p);
            v[j*4+0] = q.x; v[j*4+1] = q.y; v[j*4+2] = q.z; v[j*4+3] = q.w;
          } else {
            v[j*4+0] = 0; v[j*4+1] = 0; v[j*4+2] = 0; v[j*4+3] = 0;
          }
        }
      }
      #pragma unroll
      for (int t = 0; t < 4; ++t) {
        int acc = 0;
        #pragma unroll
        for (int k = 0; k < 7; ++k) acc += v[t + 1 + k] * w[k];
        hl[(seg * 64 + ii + t) * STR + c] = (signed char)dw_shift_clamp(acc);
      }
    }
  }

  __syncthreads();

  const int ncol   = lane & 15;
  const int hiHalf = lane >> 4;
  const int kbA    = hiHalf * 8;
  const int kbB    = hiHalf * 16;

  v8i afrag[2][2];
  #pragma unroll
  for (int mi = 0; mi < 2; ++mi) {
    const int m0 = (wave * 2 + mi) * 16;
    #pragma unroll
    for (int v = 0; v < 8; ++v) {
      const int g  = v >> 2;
      const int kA = g * 32 + kbB + (v & 3) * 4;
      afrag[mi][0][v] = *(const int*)(&wl[(m0 + ncol) * STR + kA]);
      afrag[mi][1][v] = *(const int*)(&wl[(m0 + ncol) * STR + 64 + kA]);
    }
  }

  __syncthreads();

  auto gemm_tile = [&](int n0, int mi) -> v8i {
    v8i c8 = {};
    #pragma unroll
    for (int s = 0; s < 2; ++s) {
      v8i bfrag;
      #pragma unroll
      for (int v = 0; v < 8; ++v) {
        const int grp = v >> 2;
        const int kB  = s * 64 + grp * 32 + kbB + (v & 3) * 4;
        bfrag[v] = *(const int*)(&hl[(n0 + ncol) * STR + kB]);
      }
      c8 = __builtin_amdgcn_wmma_i32_16x16x64_iu8(
                true, afrag[mi][s],
                true, bfrag, c8,
                false,  false);
    }
    return c8;
  };

  auto epilogue = [&](const v8i& c8, int n0, int m0) {
    #pragma unroll
    for (int i = 0; i < 8; ++i) {
      int r = c8[i] >> 7;
      r = r < 0 ? 0 : (r > 127 ? 127 : r);
      const int o = m0 + i + hiHalf * 8;
      stg[o * STR + n0 + ncol] = (signed char)r;
    }
  };

  #pragma unroll
  for (int mi = 0; mi < 2; ++mi) {
    const int m0 = (wave * 2 + mi) * 16;
    v8i acc[2];
    acc[0] = gemm_tile(0, mi);
    #pragma unroll
    for (int nt = 1; nt < 8; ++nt) {
      acc[nt & 1] = gemm_tile(nt * 16, mi);
      epilogue(acc[(nt - 1) & 1], (nt - 1) * 16, m0);
    }
    epilogue(acc[1], 7 * 16, m0);
  }

  __syncthreads();

  {
    const signed char* row = stg + tid * STR;
    float* orow = out + ((size_t)b * COUT + tid) * (LLEN / 2) + tile * 64;
    #pragma unroll
    for (int j = 0; j < 8; ++j) {
      const int4 q = *(const int4*)(row + j * 16);
      const int wd[4] = {q.x, q.y, q.z, q.w};
      float f[8];
      #pragma unroll
      for (int p = 0; p < 4; ++p) {
        const int b0 =  wd[p]        & 0xFF;
        const int b1 = (wd[p] >>  8) & 0xFF;
        const int b2 = (wd[p] >> 16) & 0xFF;
        const int b3 = (wd[p] >> 24) & 0xFF;
        f[p*2+0] = (float)(b0 > b1 ? b0 : b1);
        f[p*2+1] = (float)(b2 > b3 ? b2 : b3);
      }
      *(float4*)(orow + j * 8)     = make_float4(f[0], f[1], f[2], f[3]);
      *(float4*)(orow + j * 8 + 4) = make_float4(f[4], f[5], f[6], f[7]);
    }
  }
}

extern "C" void kernel_launch(void* const* d_in, const int* in_sizes, int n_in,
                              void* d_out, int out_size, void* d_ws, size_t ws_size,
                              hipStream_t stream) {
  const int* x    = (const int*)d_in[0];
  const int* w_dw = (const int*)d_in[1];
  const int* w_pw = (const int*)d_in[2];
  float* out = (float*)d_out;

  const dim3 grid(32 * 32);
  const dim3 block(256);
  hipLaunchKernelGGL(fused_dwpw_wmma_kernel, grid, block, 0, stream,
                     x, w_dw, w_pw, out);
}
